// GrailNet_28484223107669
// MI455X (gfx1250) — matrix operand order repaired and checked by reading, not compiled, not run on hardware
//
#include <hip/hip_runtime.h>
#include <math.h>

typedef __attribute__((ext_vector_type(16))) _Float16 v16h;
typedef __attribute__((ext_vector_type(8)))  float    v8f;
typedef __attribute__((ext_vector_type(4)))  float    f4;

#define NN 20000
#define BB 32
#define EE 320000
#define DD 128
#define LL 625

// ---------------------------------------------------------------------------
// WMMA helper: D = A(16x32 f16) * B(32x16 f16) + C(16x16 f32)
// ---------------------------------------------------------------------------
__device__ __forceinline__ v8f wmma32(v16h a, v16h b, v8f c) {
  return __builtin_amdgcn_wmma_f32_16x16x32_f16(false, a, false, b, (short)0, c,
                                                false, false);
}

// monotonic float<->uint encoding for atomic max
__device__ __forceinline__ unsigned fenc(float f) {
  unsigned u = __float_as_uint(f);
  return (u & 0x80000000u) ? ~u : (u | 0x80000000u);
}
__device__ __forceinline__ float fdec(unsigned u) {
  unsigned v = (u & 0x80000000u) ? (u & 0x7fffffffu) : ~u;
  return __uint_as_float(v);
}

// contiguous LDS fill (b128 copies)
__device__ __forceinline__ void lds_fill(const _Float16* g, _Float16* l,
                                         int halves, int tid) {
  const uint4* s = (const uint4*)g;
  uint4* d = (uint4*)l;
  int n = halves >> 3;
  for (int i = tid; i < n; i += 256) d[i] = s[i];
}

// gather one 16x32 A fragment row-slice for this lane (two 32B chunks) + cvt
__device__ __forceinline__ v16h load_cvt(const float* p, float s, int coff) {
  f4 f0 = *(const f4*)(p + coff);
  f4 f1 = *(const f4*)(p + coff + 4);
  f4 f2 = *(const f4*)(p + coff + 16);
  f4 f3 = *(const f4*)(p + coff + 20);
  v16h a;
  a[0] = (_Float16)(f0.x * s);  a[1] = (_Float16)(f0.y * s);
  a[2] = (_Float16)(f0.z * s);  a[3] = (_Float16)(f0.w * s);
  a[4] = (_Float16)(f1.x * s);  a[5] = (_Float16)(f1.y * s);
  a[6] = (_Float16)(f1.z * s);  a[7] = (_Float16)(f1.w * s);
  a[8] = (_Float16)(f2.x * s);  a[9] = (_Float16)(f2.y * s);
  a[10] = (_Float16)(f2.z * s); a[11] = (_Float16)(f2.w * s);
  a[12] = (_Float16)(f3.x * s); a[13] = (_Float16)(f3.y * s);
  a[14] = (_Float16)(f3.z * s); a[15] = (_Float16)(f3.w * s);
  return a;
}

struct Tile {
  const float* p[4];
  float s[4];
};

// ---------------------------------------------------------------------------
// Weight swizzle: W[K][128] f32 -> B-fragment-ordered f16.
// Fragment (kk,n): lane c holds col N=n*16+(c%16), halves j -> K=kk*32+(c/16)*8+j+(j/8)*8
// ---------------------------------------------------------------------------
extern "C" __global__ void k_swizzle(const float* __restrict__ W,
                                     _Float16* __restrict__ dst, int nK) {
  int idx = blockIdx.x * 256 + threadIdx.x;
  if (idx >= nK * 256) return;
  int lane = idx & 31, frag = idx >> 5;
  int n = frag & 7, kk = frag >> 3;
  int Nc = n * 16 + (lane & 15);
  int Kb = kk * 32 + (lane >> 4) * 8;
  _Float16* d = dst + (size_t)idx * 16;
#pragma unroll
  for (int j = 0; j < 16; ++j) d[j] = (_Float16)W[(Kb + j + (j >> 3) * 8) * 128 + Nc];
}

// ---------------------------------------------------------------------------
// Core: per-wave DUAL 16-row tiles (32 rows), 2-layer MLP.
// NK1 is a compile-time constant so source selects are constant-index ->
// address-space inference succeeds -> global_load_b128 (not flat).
// Each LDS B fragment feeds two WMMAs (halves ds traffic per matrix op).
// ---------------------------------------------------------------------------
template <int NK1>
__device__ __forceinline__ void mlp2_x2(Tile t0, Tile t1,
                                        const _Float16* ldsW1,
                                        const _Float16* ldsW2,
                                        _Float16* hbuf /*32x128 halves*/,
                                        const float* b1, int lane, v8f c2a[8],
                                        v8f c2b[8]) {
  v8f z = {};
  v8f ca[8], cb[8];
#pragma unroll
  for (int n = 0; n < 8; ++n) { ca[n] = z; cb[n] = z; }

#pragma unroll
  for (int kk = 0; kk < NK1; ++kk) {
    const int si = kk >> 2;
    int coff = ((kk & 3) << 5) + ((lane >> 4) << 3);
    v16h a0 = load_cvt(t0.p[si], t0.s[si], coff);
    v16h a1 = load_cvt(t1.p[si], t1.s[si], coff);
    const v16h* bw = (const v16h*)ldsW1 + (kk << 3) * 32 + lane;
#pragma unroll
    for (int n = 0; n < 8; ++n) {
      v16h b = bw[n * 32];
      ca[n] = wmma32(a0, b, ca[n]);
      cb[n] = wmma32(a1, b, cb[n]);
    }
  }

  // bias + relu -> f16 staging tile (C layout -> row-major), rows 0-15 / 16-31
  int nco = lane & 15, cg = lane >> 4;
#pragma unroll
  for (int n = 0; n < 8; ++n) {
    float bias = b1[n * 16 + nco];
#pragma unroll
    for (int r = 0; r < 8; ++r) {
      float h0 = ca[n][r] + bias;
      float h1 = cb[n][r] + bias;
      h0 = h0 > 0.f ? h0 : 0.f;
      h1 = h1 > 0.f ? h1 : 0.f;
      hbuf[(r + cg * 8) * 128 + n * 16 + nco] = (_Float16)h0;
      hbuf[(16 + r + cg * 8) * 128 + n * 16 + nco] = (_Float16)h1;
    }
  }
  __syncthreads();

#pragma unroll
  for (int n = 0; n < 8; ++n) { c2a[n] = z; c2b[n] = z; }
#pragma unroll
  for (int kk = 0; kk < 4; ++kk) {
    const _Float16* hp0 =
        hbuf + (lane & 15) * 128 + (kk << 5) + ((lane >> 4) << 3);
    const _Float16* hp1 = hp0 + 16 * 128;
    v16h a0, a1;
#pragma unroll
    for (int j = 0; j < 8; ++j) {
      a0[j] = hp0[j]; a0[8 + j] = hp0[16 + j];
      a1[j] = hp1[j]; a1[8 + j] = hp1[16 + j];
    }
    const v16h* bw = (const v16h*)ldsW2 + (kk << 3) * 32 + lane;
#pragma unroll
    for (int n = 0; n < 8; ++n) {
      v16h b = bw[n * 32];
      c2a[n] = wmma32(a0, b, c2a[n]);
      c2b[n] = wmma32(a1, b, c2b[n]);
    }
  }
}

// ---------------------------------------------------------------------------
// Edge MLP: A = [x[row], x[col], edge_attr, u[eb]] (K=512) -> 128 -> 128
// block = 8 waves x 32 edges = 256 edges
// ---------------------------------------------------------------------------
extern "C" __global__ void __launch_bounds__(256)
k_edge_mlp(const float* __restrict__ x, const float* __restrict__ ea,
           const float* __restrict__ u, const int* __restrict__ ei,
           const int* __restrict__ nb, const _Float16* __restrict__ W1,
           const float* __restrict__ b1, const _Float16* __restrict__ W2,
           const float* __restrict__ b2, float* __restrict__ eout) {
  extern __shared__ char smem[];
  _Float16* lW1 = (_Float16*)smem;
  _Float16* lW2 = lW1 + 16 * 4096;
  _Float16* hb = lW2 + 4 * 4096;
  int tid = threadIdx.x;
  lds_fill(W1, lW1, 16 * 4096, tid);
  lds_fill(W2, lW2, 4 * 4096, tid);
  __syncthreads();
  int wave = tid >> 5, lane = tid & 31;
  long e32 = (long)blockIdx.x * 256 + wave * 32;
  long e0 = e32 + (lane & 15);
  long e1 = e0 + 16;
  int r0 = ei[e0], c0 = ei[EE + e0], g0 = nb[r0];
  int r1 = ei[e1], c1 = ei[EE + e1], g1 = nb[r1];
  Tile t0 = {{x + (long)r0 * 128, x + (long)c0 * 128, ea + e0 * 128,
              u + (long)g0 * 128},
             {1.f, 1.f, 1.f, 1.f}};
  Tile t1 = {{x + (long)r1 * 128, x + (long)c1 * 128, ea + e1 * 128,
              u + (long)g1 * 128},
             {1.f, 1.f, 1.f, 1.f}};
  v8f c2a[8], c2b[8];
  mlp2_x2<16>(t0, t1, lW1, lW2, hb + wave * 4096, b1, lane, c2a, c2b);
  int nco = lane & 15, cg = lane >> 4;
#pragma unroll
  for (int n = 0; n < 8; ++n) {
    float bias = b2[n * 16 + nco];
#pragma unroll
    for (int r = 0; r < 8; ++r) {
      eout[(e32 + r + cg * 8) * 128 + n * 16 + nco] = c2a[n][r] + bias;
      eout[(e32 + 16 + r + cg * 8) * 128 + n * 16 + nco] = c2b[n][r] + bias;
    }
  }
}

// ---------------------------------------------------------------------------
// Message MLP: A = [x[row], new_edge] (K=256) -> 128 -> 128, then
// scatter msg*norm_w into summed[col] with f32 atomics.
// ---------------------------------------------------------------------------
extern "C" __global__ void __launch_bounds__(256)
k_msg(const float* __restrict__ x, const float* __restrict__ ne,
      const int* __restrict__ ei, const _Float16* __restrict__ W1,
      const float* __restrict__ b1, const _Float16* __restrict__ W2,
      const float* __restrict__ b2, const float* __restrict__ normw,
      float* __restrict__ summed) {
  extern __shared__ char smem[];
  _Float16* lW1 = (_Float16*)smem;
  _Float16* lW2 = lW1 + 8 * 4096;
  _Float16* hb = lW2 + 4 * 4096;
  int tid = threadIdx.x;
  lds_fill(W1, lW1, 8 * 4096, tid);
  lds_fill(W2, lW2, 4 * 4096, tid);
  __syncthreads();
  int wave = tid >> 5, lane = tid & 31;
  long e32 = (long)blockIdx.x * 256 + wave * 32;
  long e0 = e32 + (lane & 15);
  long e1 = e0 + 16;
  int r0 = ei[e0], r1 = ei[e1];
  Tile t0 = {{x + (long)r0 * 128, ne + e0 * 128, x + (long)r0 * 128,
              x + (long)r0 * 128},
             {1.f, 1.f, 1.f, 1.f}};
  Tile t1 = {{x + (long)r1 * 128, ne + e1 * 128, x + (long)r1 * 128,
              x + (long)r1 * 128},
             {1.f, 1.f, 1.f, 1.f}};
  v8f c2a[8], c2b[8];
  mlp2_x2<8>(t0, t1, lW1, lW2, hb + wave * 4096, b1, lane, c2a, c2b);
  int nco = lane & 15, cg = lane >> 4;
  float wgtA[8], wgtB[8];
  int dstA[8], dstB[8];
#pragma unroll
  for (int r = 0; r < 8; ++r) {
    long ea2 = e32 + r + cg * 8;
    long eb2 = ea2 + 16;
    wgtA[r] = normw[ea2];
    dstA[r] = ei[EE + ea2];
    wgtB[r] = normw[eb2];
    dstB[r] = ei[EE + eb2];
  }
#pragma unroll
  for (int n = 0; n < 8; ++n) {
    float bias = b2[n * 16 + nco];
#pragma unroll
    for (int r = 0; r < 8; ++r) {
      atomicAdd(&summed[(long)dstA[r] * 128 + n * 16 + nco],
                (c2a[n][r] + bias) * wgtA[r]);
      atomicAdd(&summed[(long)dstB[r] * 128 + n * 16 + nco],
                (c2b[n][r] + bias) * wgtB[r]);
    }
  }
}

// ---------------------------------------------------------------------------
// Node update MLP: A = [x, summed*inv_cnt, u[nb]] (K=384) -> 128 -> 128
// ---------------------------------------------------------------------------
extern "C" __global__ void __launch_bounds__(256)
k_update(const float* __restrict__ x, const float* __restrict__ summed,
         const float* __restrict__ invc, const float* __restrict__ u,
         const int* __restrict__ nb, const _Float16* __restrict__ W1,
         const float* __restrict__ b1, const _Float16* __restrict__ W2,
         const float* __restrict__ b2, float* __restrict__ xout) {
  extern __shared__ char smem[];
  _Float16* lW1 = (_Float16*)smem;
  _Float16* lW2 = lW1 + 12 * 4096;
  _Float16* hb = lW2 + 4 * 4096;
  int tid = threadIdx.x;
  lds_fill(W1, lW1, 12 * 4096, tid);
  lds_fill(W2, lW2, 4 * 4096, tid);
  __syncthreads();
  int wave = tid >> 5, lane = tid & 31;
  long n32 = (long)blockIdx.x * 256 + wave * 32;
  long nd0 = n32 + (lane & 15);
  long nd1 = nd0 + 16;
  long nc0 = nd0 < NN ? nd0 : (long)NN - 1;
  long nc1 = nd1 < NN ? nd1 : (long)NN - 1;
  Tile t0 = {{x + nc0 * 128, summed + nc0 * 128, u + (long)nb[nc0] * 128,
              x + nc0 * 128},
             {1.f, invc[nc0], 1.f, 1.f}};
  Tile t1 = {{x + nc1 * 128, summed + nc1 * 128, u + (long)nb[nc1] * 128,
              x + nc1 * 128},
             {1.f, invc[nc1], 1.f, 1.f}};
  v8f c2a[8], c2b[8];
  mlp2_x2<12>(t0, t1, lW1, lW2, hb + wave * 4096, b1, lane, c2a, c2b);
  int nco = lane & 15, cg = lane >> 4;
#pragma unroll
  for (int n = 0; n < 8; ++n) {
    float bias = b2[n * 16 + nco];
#pragma unroll
    for (int r = 0; r < 8; ++r) {
      long na = n32 + r + cg * 8;
      long nbr = na + 16;
      if (na < NN) xout[na * 128 + n * 16 + nco] = c2a[n][r] + bias;
      if (nbr < NN) xout[nbr * 128 + n * 16 + nco] = c2b[n][r] + bias;
    }
  }
}

// ---------------------------------------------------------------------------
// K/V projections of evidence (single-layer linears, shared A fragments)
// ---------------------------------------------------------------------------
extern "C" __global__ void __launch_bounds__(256)
k_kv(const float* __restrict__ ev, const _Float16* __restrict__ Wk,
     const float* __restrict__ bk, const _Float16* __restrict__ Wv,
     const float* __restrict__ bv, float* __restrict__ kout,
     float* __restrict__ vout) {
  extern __shared__ char smem[];
  _Float16* lWk = (_Float16*)smem;
  _Float16* lWv = lWk + 4 * 4096;
  int tid = threadIdx.x;
  lds_fill(Wk, lWk, 4 * 4096, tid);
  lds_fill(Wv, lWv, 4 * 4096, tid);
  __syncthreads();
  int wave = tid >> 5, lane = tid & 31;
  long n16 = (long)blockIdx.x * 128 + wave * 16;
  long node = n16 + (lane & 15);
  long ncl = node < NN ? node : (long)NN - 1;
  const float* p = ev + ncl * 128;
  v16h afr[4];
#pragma unroll
  for (int kk = 0; kk < 4; ++kk) {
    int coff = (kk << 5) + ((lane >> 4) << 3);
    afr[kk] = load_cvt(p, 1.f, coff);
  }
  v8f zero = {};
  v8f ck[8], cv[8];
#pragma unroll
  for (int n = 0; n < 8; ++n) { ck[n] = zero; cv[n] = zero; }
#pragma unroll
  for (int kk = 0; kk < 4; ++kk) {
    const v16h* bkp = (const v16h*)lWk + (kk << 3) * 32 + lane;
    const v16h* bvp = (const v16h*)lWv + (kk << 3) * 32 + lane;
#pragma unroll
    for (int n = 0; n < 8; ++n) {
      ck[n] = wmma32(afr[kk], bkp[n * 32], ck[n]);
      cv[n] = wmma32(afr[kk], bvp[n * 32], cv[n]);
    }
  }
  int nco = lane & 15, cg = lane >> 4;
#pragma unroll
  for (int n = 0; n < 8; ++n) {
    float bbk = bk[n * 16 + nco], bbv = bv[n * 16 + nco];
#pragma unroll
    for (int r = 0; r < 8; ++r) {
      long nd = n16 + r + cg * 8;
      if (nd < NN) {
        kout[nd * 128 + n * 16 + nco] = ck[n][r] + bbk;
        vout[nd * 128 + n * 16 + nco] = cv[n][r] + bbv;
      }
    }
  }
}

// ---------------------------------------------------------------------------
// Small kernels
// ---------------------------------------------------------------------------
extern "C" __global__ void __launch_bounds__(256)
k_logit(const float* __restrict__ ea, const float* __restrict__ u,
        const int* __restrict__ ei, const int* __restrict__ nb,
        const float* __restrict__ Ww, const float* __restrict__ bw,
        float* __restrict__ wts) {
  int wave = threadIdx.x >> 5, lane = threadIdx.x & 31;
  long e = (long)blockIdx.x * 8 + wave;
  if (e >= EE) return;
  int r0 = ei[e];
  int g = nb[r0];
  const float* pe = ea + e * 128;
  const float* pu = u + (long)g * 128;
  float s = 0.f;
#pragma unroll
  for (int j = 0; j < 4; ++j) {
    int i = lane + j * 32;
    s += pe[i] * Ww[i] + pu[i] * Ww[128 + i];
  }
#pragma unroll
  for (int off = 16; off > 0; off >>= 1) s += __shfl_xor(s, off, 32);
  if (lane == 0) wts[e] = s + bw[0];
}

extern "C" __global__ void k_segmax(const float* __restrict__ wts,
                                    const int* __restrict__ ei,
                                    const int* __restrict__ nb,
                                    unsigned* __restrict__ segmax) {
  long e = (long)blockIdx.x * 256 + threadIdx.x;
  if (e >= EE) return;
  atomicMax(&segmax[nb[ei[e]]], fenc(wts[e]));
}

extern "C" __global__ void k_expsum(const float* __restrict__ wts,
                                    const int* __restrict__ ei,
                                    const int* __restrict__ nb,
                                    const unsigned* __restrict__ segmax,
                                    float* __restrict__ segsum,
                                    float* __restrict__ normw) {
  long e = (long)blockIdx.x * 256 + threadIdx.x;
  if (e >= EE) return;
  int g = nb[ei[e]];
  float ex = expf(wts[e] - fdec(segmax[g]));
  normw[e] = ex;
  atomicAdd(&segsum[g], ex);
}

extern "C" __global__ void k_normdiv(float* __restrict__ normw,
                                     const int* __restrict__ ei,
                                     const int* __restrict__ nb,
                                     const float* __restrict__ segsum) {
  long e = (long)blockIdx.x * 256 + threadIdx.x;
  if (e >= EE) return;
  normw[e] /= segsum[nb[ei[e]]];
}

extern "C" __global__ void k_cnt(const int* __restrict__ ei,
                                 int* __restrict__ cnt) {
  long e = (long)blockIdx.x * 256 + threadIdx.x;
  if (e >= EE) return;
  atomicAdd(&cnt[ei[EE + e]], 1);
}

extern "C" __global__ void k_inv(const int* __restrict__ cnt,
                                 float* __restrict__ invc) {
  int n = blockIdx.x * 256 + threadIdx.x;
  if (n >= NN) return;
  invc[n] = 1.f / fmaxf((float)cnt[n], 1.f);
}

extern "C" __global__ void k_gelu(const float* __restrict__ x,
                                  float* __restrict__ ev) {
  long i = (long)blockIdx.x * 256 + threadIdx.x;
  if (i >= (long)NN * 128) return;
  float v = x[i];
  ev[i] = 0.5f * v * (1.f + erff(v * 0.70710678118654752f));
}

extern "C" __global__ void k_q(const float* __restrict__ qa,
                               const float* __restrict__ Wq,
                               const float* __restrict__ bq,
                               float* __restrict__ qbuf) {
  int b = blockIdx.x, f = threadIdx.x;  // 128 threads
  float acc = bq[f];
  const float* qr = qa + (long)b * 1024;
  for (int i = 0; i < 1024; ++i) acc += qr[i] * Wq[i * 128 + f];
  qbuf[b * 128 + f] = acc;
}

extern "C" __global__ void __launch_bounds__(256)
k_pooled_edge(const float* __restrict__ ea, const float* __restrict__ normw,
              const int* __restrict__ ei, const int* __restrict__ nb,
              float* __restrict__ out) {
  __shared__ float acc[BB * DD];
  int tid = threadIdx.x;
  for (int i = tid; i < BB * DD; i += 256) acc[i] = 0.f;
  __syncthreads();
  const long EPB = EE / 256;  // grid = 256 blocks
  long e0 = (long)blockIdx.x * EPB;
  int f = tid & 127, sub = tid >> 7;
  for (long e = e0 + sub; e < e0 + EPB; e += 2) {
    int g = nb[ei[e]];
    atomicAdd(&acc[g * DD + f], ea[e * 128 + f] * normw[e]);
  }
  __syncthreads();
  for (int i = tid; i < BB * DD; i += 256)
    atomicAdd(&out[(i >> 7) * 256 + (i & 127)], acc[i]);
}

extern "C" __global__ void __launch_bounds__(256)
k_attn(const float* __restrict__ qbuf, const float* __restrict__ kbuf,
       const float* __restrict__ vbuf, const int* __restrict__ non,
       float* __restrict__ out) {
  __shared__ float sc[LL];
  __shared__ float red[256];
  __shared__ float qsh[64];
  int b = blockIdx.x >> 1, h = blockIdx.x & 1;
  int tid = threadIdx.x;
  if (tid < 64) qsh[tid] = qbuf[b * 128 + h * 64 + tid];
  __syncthreads();
  int lim = non[b];
  float lmax = -3.0e38f;
  for (int l = tid; l < LL; l += 256) {
    const float* kp = kbuf + ((long)(b * LL + l)) * 128 + h * 64;
    float s = 0.f;
#pragma unroll
    for (int d = 0; d < 64; ++d) s += qsh[d] * kp[d];
    s *= 0.125f;  // 1/sqrt(64)
    if (l >= lim) s = -1.0e30f;
    sc[l] = s;
    lmax = fmaxf(lmax, s);
  }
  red[tid] = lmax;
  __syncthreads();
  for (int s2 = 128; s2 > 0; s2 >>= 1) {
    if (tid < s2) red[tid] = fmaxf(red[tid], red[tid + s2]);
    __syncthreads();
  }
  float m = red[0];
  __syncthreads();
  float ls = 0.f;
  for (int l = tid; l < LL; l += 256) {
    float e2 = expf(sc[l] - m);
    sc[l] = e2;
    ls += e2;
  }
  red[tid] = ls;
  __syncthreads();
  for (int s2 = 128; s2 > 0; s2 >>= 1) {
    if (tid < s2) red[tid] += red[tid + s2];
    __syncthreads();
  }
  float invS = 1.f / red[0];
  __syncthreads();
  int d = tid & 63, grp = tid >> 6;
  float acc = 0.f;
  for (int l = grp; l < LL; l += 4)
    acc += sc[l] * vbuf[((long)(b * LL + l)) * 128 + h * 64 + d];
  red[tid] = acc;
  __syncthreads();
  if (grp == 0)
    out[b * 256 + 128 + h * 64 + d] =
        (red[d] + red[64 + d] + red[128 + d] + red[192 + d]) * invS;
}

// ---------------------------------------------------------------------------
// Host driver
// ---------------------------------------------------------------------------
extern "C" void kernel_launch(void* const* d_in, const int* in_sizes, int n_in,
                              void* d_out, int out_size, void* d_ws,
                              size_t ws_size, hipStream_t stream) {
  (void)in_sizes; (void)n_in; (void)ws_size;
  const float* x = (const float*)d_in[0];
  const float* ea = (const float*)d_in[1];
  const float* u = (const float*)d_in[2];
  const float* qa = (const float*)d_in[3];
  const float* We1 = (const float*)d_in[4];  const float* be1 = (const float*)d_in[5];
  const float* We2 = (const float*)d_in[6];  const float* be2 = (const float*)d_in[7];
  const float* Ww  = (const float*)d_in[8];  const float* bw  = (const float*)d_in[9];
  const float* Wm1 = (const float*)d_in[10]; const float* bm1 = (const float*)d_in[11];
  const float* Wm2 = (const float*)d_in[12]; const float* bm2 = (const float*)d_in[13];
  const float* Wu1 = (const float*)d_in[14]; const float* bu1 = (const float*)d_in[15];
  const float* Wu2 = (const float*)d_in[16]; const float* bu2 = (const float*)d_in[17];
  const float* Wq  = (const float*)d_in[18]; const float* bq  = (const float*)d_in[19];
  const float* Wk  = (const float*)d_in[20]; const float* bk  = (const float*)d_in[21];
  const float* Wv  = (const float*)d_in[22]; const float* bv  = (const float*)d_in[23];
  const int* ei  = (const int*)d_in[24];
  const int* nb  = (const int*)d_in[25];
  const int* non = (const int*)d_in[26];
  float* out = (float*)d_out;

  char* w = (char*)d_ws;
  auto alloc = [&](size_t bytes) -> void* {
    void* p = w;
    w += (bytes + 255) & ~(size_t)255;
    return p;
  };
  _Float16* We1h = (_Float16*)alloc((size_t)16 * 4096 * 2);
  _Float16* We2h = (_Float16*)alloc((size_t)4 * 4096 * 2);
  _Float16* Wm1h = (_Float16*)alloc((size_t)8 * 4096 * 2);
  _Float16* Wm2h = (_Float16*)alloc((size_t)4 * 4096 * 2);
  _Float16* Wu1h = (_Float16*)alloc((size_t)12 * 4096 * 2);
  _Float16* Wu2h = (_Float16*)alloc((size_t)4 * 4096 * 2);
  _Float16* Wkh  = (_Float16*)alloc((size_t)4 * 4096 * 2);
  _Float16* Wvh  = (_Float16*)alloc((size_t)4 * 4096 * 2);
  float* eA     = (float*)alloc((size_t)EE * 128 * 4);
  float* eB     = (float*)alloc((size_t)EE * 128 * 4);
  float* xA     = (float*)alloc((size_t)NN * 128 * 4);
  float* xB     = (float*)alloc((size_t)NN * 128 * 4);
  float* summed = (float*)alloc((size_t)NN * 128 * 4);
  float* evid   = (float*)alloc((size_t)NN * 128 * 4);
  float* kbuf   = (float*)alloc((size_t)NN * 128 * 4);
  float* vbuf   = (float*)alloc((size_t)NN * 128 * 4);
  float* wts    = (float*)alloc((size_t)EE * 4);
  float* normw  = (float*)alloc((size_t)EE * 4);
  float* qbuf   = (float*)alloc((size_t)BB * 128 * 4);
  unsigned* segmax = (unsigned*)alloc(BB * 4);
  float* segsum    = (float*)alloc(BB * 4);
  int* cnt   = (int*)alloc((size_t)NN * 4);
  float* invc = (float*)alloc((size_t)NN * 4);

  const int EDGE_SMEM = 16 * 8192 + 32768 + 65536;  // 229376
  const int MSG_SMEM  = 8 * 8192 + 32768 + 65536;   // 163840
  const int UPD_SMEM  = 12 * 8192 + 32768 + 65536;  // 196608
  const int KV_SMEM   = 8 * 4096 * 2;               // 65536
  hipFuncSetAttribute((const void*)k_edge_mlp,
                      hipFuncAttributeMaxDynamicSharedMemorySize, EDGE_SMEM);
  hipFuncSetAttribute((const void*)k_msg,
                      hipFuncAttributeMaxDynamicSharedMemorySize, MSG_SMEM);
  hipFuncSetAttribute((const void*)k_update,
                      hipFuncAttributeMaxDynamicSharedMemorySize, UPD_SMEM);
  hipFuncSetAttribute((const void*)k_kv,
                      hipFuncAttributeMaxDynamicSharedMemorySize, KV_SMEM);

  // weight swizzles (once per launch)
  k_swizzle<<<16, 256, 0, stream>>>(We1, We1h, 16);
  k_swizzle<<<4, 256, 0, stream>>>(We2, We2h, 4);
  k_swizzle<<<8, 256, 0, stream>>>(Wm1, Wm1h, 8);
  k_swizzle<<<4, 256, 0, stream>>>(Wm2, Wm2h, 4);
  k_swizzle<<<12, 256, 0, stream>>>(Wu1, Wu1h, 12);
  k_swizzle<<<4, 256, 0, stream>>>(Wu2, Wu2h, 4);
  k_swizzle<<<4, 256, 0, stream>>>(Wk, Wkh, 4);
  k_swizzle<<<4, 256, 0, stream>>>(Wv, Wvh, 4);

  hipMemsetAsync(out, 0, (size_t)out_size * 4, stream);
  hipMemsetAsync(cnt, 0, (size_t)NN * 4, stream);
  k_cnt<<<(EE + 255) / 256, 256, 0, stream>>>(ei, cnt);
  k_inv<<<(NN + 255) / 256, 256, 0, stream>>>(cnt, invc);
  k_q<<<BB, 128, 0, stream>>>(qa, Wq, bq, qbuf);

  const float* cx = x;
  const float* ce = ea;
  float* ebufs[2] = {eA, eB};
  float* xbufs[2] = {xA, xB};
  for (int layer = 0; layer < 3; ++layer) {
    float* ne = ebufs[layer & 1];
    float* nx = xbufs[layer & 1];
    hipMemsetAsync(segmax, 0, BB * 4, stream);
    hipMemsetAsync(segsum, 0, BB * 4, stream);
    hipMemsetAsync(summed, 0, (size_t)NN * 128 * 4, stream);
    k_logit<<<EE / 8, 256, 0, stream>>>(ce, u, ei, nb, Ww, bw, wts);
    k_segmax<<<(EE + 255) / 256, 256, 0, stream>>>(wts, ei, nb, segmax);
    k_expsum<<<(EE + 255) / 256, 256, 0, stream>>>(wts, ei, nb, segmax, segsum,
                                                   normw);
    k_normdiv<<<(EE + 255) / 256, 256, 0, stream>>>(normw, ei, nb, segsum);
    k_edge_mlp<<<EE / 256, 256, EDGE_SMEM, stream>>>(cx, ce, u, ei, nb, We1h,
                                                     be1, We2h, be2, ne);
    k_msg<<<EE / 256, 256, MSG_SMEM, stream>>>(cx, ne, ei, Wm1h, bm1, Wm2h,
                                               bm2, normw, summed);
    k_update<<<(NN + 255) / 256, 256, UPD_SMEM, stream>>>(
        cx, summed, invc, u, nb, Wu1h, bu1, Wu2h, bu2, nx);
    cx = nx;
    ce = ne;
  }

  k_pooled_edge<<<256, 256, 0, stream>>>(ce, normw, ei, nb, out);
  k_gelu<<<((long)NN * 128 + 255) / 256, 256, 0, stream>>>(cx, evid);
  k_kv<<<(NN + 127) / 128, 256, KV_SMEM, stream>>>(evid, Wkh, bk, Wvh, bv,
                                                   kbuf, vbuf);
  k_attn<<<BB * 2, 256, 0, stream>>>(qbuf, kbuf, vbuf, non, out);
}
